// SuperGlue_22582938043095
// MI455X (gfx1250) — compile-verified, matrix operand order checked by reading, not run on hardware
//
#include <hip/hip_runtime.h>
#include <hip/hip_bf16.h>

typedef __attribute__((ext_vector_type(16))) _Float16 v16h;
typedef __attribute__((ext_vector_type(8)))  float    v8f;

#define NKP 1024
#define DIM 256
#define MP  (NKP + 1)

// ---------------- WMMA GEMM: C[M,N] = act(alpha*A[M,K]*B[K,N] + bias, BN, relu) + resid ----
#define BM 64
#define BK 32
#define LDS_PITCH (BK + 8)

// Fragment loader following CDNA5 16-bit A-matrix layout (CDNA5 ISA):
// lane%16 -> row within 16-row tile, lane/16 selects K-halves {0..7,16..23} vs {8..15,24..31}.
// Element j maps to k = (j/8)*16 + (lane/16)*8 + (j%8). B is staged transposed in LDS so
// the same loader serves both operands (contiguous 8-element groups -> ds_load_b128 x2).
__device__ __forceinline__ v16h frag_load(const _Float16* base, int row0, int lane) {
  const _Float16* p = base + (size_t)(row0 + (lane & 15)) * LDS_PITCH + ((lane >> 4) * 8);
  v16h f;
#pragma unroll
  for (int j = 0; j < 16; ++j) {
    int k = ((j >> 3) * 16) + (j & 7);
    f[j] = p[k];
  }
  return f;
}

// TBN = 128: 8 waves as 4 row-tiles x 2 col-groups, 4 col-tiles (4 WMMA) per wave.
// TBN =  64: same arrangement, 2 col-tiles per wave. Used when N is not a multiple of 128.
template <int TBN>
__global__ void __launch_bounds__(256) gemm_wmma_kernel(
    const float* __restrict__ A, int lda,
    const float* __restrict__ B, int ldb,
    float* __restrict__ C, int ldc,
    int M, int N, int K, float alpha,
    const float* __restrict__ bias,
    const float* __restrict__ bng, const float* __restrict__ bnb,
    const float* __restrict__ bnm, const float* __restrict__ bnv,
    const float* __restrict__ resid, int ldr, int relu)
{
  constexpr int CPW = TBN / 32;              // col tiles per wave (2 col groups of waves)
  __shared__ _Float16 sA[BM * LDS_PITCH];    // [BM][BK]
  __shared__ _Float16 sBT[TBN * LDS_PITCH];  // transposed: [TBN][BK]

  const int tid  = threadIdx.x;
  const int lane = tid & 31;
  const int wave = tid >> 5;        // 8 waves
  const int wr   = wave & 3;        // row tile 0..3
  const int wg   = wave >> 2;       // col group 0..1
  const int bm   = blockIdx.y * BM;
  const int bn   = blockIdx.x * TBN;

  v8f acc[CPW];
  const v8f zero = {0.f, 0.f, 0.f, 0.f, 0.f, 0.f, 0.f, 0.f};
#pragma unroll
  for (int i = 0; i < CPW; ++i) acc[i] = zero;

  const bool aligned  = ((lda & 3) == 0) && ((ldb & 3) == 0);
  const bool interior = (bm + BM <= M) && (bn + TBN <= N);

  for (int k0 = 0; k0 < K; k0 += BK) {
    const bool full = aligned && interior && (k0 + BK <= K);
    if (full) {
      // prefetch next K-step tiles into L2 while we work on this one
      if (k0 + BK < K) {
        __builtin_prefetch(A + (size_t)(bm + (tid >> 3)) * lda + (k0 + BK), 0, 1);
        __builtin_prefetch(B + (size_t)(k0 + BK + (tid >> 3)) * ldb + bn, 0, 1);
      }
      // A tile: BM*BK floats as float4 (global_load_b128), contiguous f16 stores
#pragma unroll
      for (int it = 0; it < (BM * BK) / 1024; ++it) {
        int idx = tid + it * 256;            // over BM*BK/4
        int r = idx >> 3, c4 = (idx & 7) * 4;
        float4 v = *(const float4*)(A + (size_t)(bm + r) * lda + (k0 + c4));
        _Float16* d = &sA[r * LDS_PITCH + c4];
        d[0] = (_Float16)v.x; d[1] = (_Float16)v.y;
        d[2] = (_Float16)v.z; d[3] = (_Float16)v.w;
      }
      // B tile: BK*TBN floats, stored transposed
#pragma unroll
      for (int it = 0; it < (BK * TBN) / 1024; ++it) {
        int idx = tid + it * 256;            // over BK*TBN/4
        int r = idx / (TBN / 4), c4 = (idx % (TBN / 4)) * 4;
        float4 v = *(const float4*)(B + (size_t)(k0 + r) * ldb + (bn + c4));
        sBT[(c4 + 0) * LDS_PITCH + r] = (_Float16)v.x;
        sBT[(c4 + 1) * LDS_PITCH + r] = (_Float16)v.y;
        sBT[(c4 + 2) * LDS_PITCH + r] = (_Float16)v.z;
        sBT[(c4 + 3) * LDS_PITCH + r] = (_Float16)v.w;
      }
    } else {
      // ragged tile: branch-free (clamped address + select), no exec divergence
#pragma unroll
      for (int it = 0; it < (BM * BK) / 256; ++it) {
        int t = tid + it * 256;
        int r = t >> 5, c = t & 31;
        int gr = bm + r, gc = k0 + c;
        bool ok = (gr < M) && (gc < K);
        size_t idx = ok ? ((size_t)gr * lda + gc) : 0;
        float v = A[idx];
        sA[r * LDS_PITCH + c] = ok ? (_Float16)v : (_Float16)0.f;
      }
#pragma unroll
      for (int it = 0; it < (BK * TBN) / 256; ++it) {
        int t = tid + it * 256;
        int r = t / TBN, c = t % TBN;
        int gr = k0 + r, gc = bn + c;
        bool ok = (gr < K) && (gc < N);
        size_t idx = ok ? ((size_t)gr * ldb + gc) : 0;
        float v = B[idx];
        sBT[c * LDS_PITCH + r] = ok ? (_Float16)v : (_Float16)0.f;
      }
    }
    __syncthreads();
    v16h fa = frag_load(sA, wr * 16, lane);
#pragma unroll
    for (int cc = 0; cc < CPW; ++cc) {
      v16h fb = frag_load(sBT, (wg * CPW + cc) * 16, lane);
      acc[cc] = __builtin_amdgcn_wmma_f32_16x16x32_f16(false, fa, false, fb,
                                                       (short)0, acc[cc], false, false);
    }
    __syncthreads();
  }

  // C/D layout: VGPR r -> M = r + 8*(lane/16); N = lane%16
  const int ncol  = lane & 15;
  const int mhalf = (lane >> 4) * 8;
  if (interior) {
#pragma unroll
    for (int cc = 0; cc < CPW; ++cc) {
      int n = bn + (wg * CPW + cc) * 16 + ncol;
#pragma unroll
      for (int r = 0; r < 8; ++r) {
        int m = bm + wr * 16 + mhalf + r;
        float y = acc[cc][r] * alpha;
        if (bias) y += bias[m];
        if (bng)  y = (y - bnm[m]) * rsqrtf(bnv[m] + 1e-5f) * bng[m] + bnb[m];
        if (relu) y = fmaxf(y, 0.f);
        if (resid) y += resid[(size_t)m * ldr + n];
        C[(size_t)m * ldc + n] = y;
      }
    }
  } else {
#pragma unroll
    for (int cc = 0; cc < CPW; ++cc) {
      int n = bn + (wg * CPW + cc) * 16 + ncol;
#pragma unroll
      for (int r = 0; r < 8; ++r) {
        int m = bm + wr * 16 + mhalf + r;
        if (m < M && n < N) {
          float y = acc[cc][r] * alpha;
          if (bias) y += bias[m];
          if (bng)  y = (y - bnm[m]) * rsqrtf(bnv[m] + 1e-5f) * bng[m] + bnb[m];
          if (relu) y = fmaxf(y, 0.f);
          if (resid) y += resid[(size_t)m * ldr + n];
          C[(size_t)m * ldc + n] = y;
        }
      }
    }
  }
}

// ---------------- keypoint-encoder input: [normalized kx; ky; score] (3 x 1024) ----------
__global__ void build_input_kernel(const float* __restrict__ kpts,
                                   const float* __restrict__ scores,
                                   const int* __restrict__ wptr, const int* __restrict__ hptr,
                                   float* __restrict__ out)
{
  int n = blockIdx.x * 256 + threadIdx.x;
  if (n >= NKP) return;
  float w = (float)wptr[0], h = (float)hptr[0];
  float cx = w * 0.5f, cy = h * 0.5f;
  float sc = fmaxf(w, h) * 0.7f;
  out[0 * NKP + n] = (kpts[n * 2 + 0] - cx) / sc;
  out[1 * NKP + n] = (kpts[n * 2 + 1] - cy) / sc;
  out[2 * NKP + n] = scores[n];
}

// ---------------- head split/merge: channel c = hd*4 + h (reshape(b, 64, 4, N)) ----------
// mode 0: dst[h][n][64] = src[c][n]  (Q^T / V^T per head, row-major n x 64)
// mode 1: dst[h][64][N] = src[c][n]  (K per head, row-major 64 x N)
// mode 2: dst[c][n] = src[h][n][64]  (merge heads back to d-major)
__global__ void headshuffle_kernel(const float* __restrict__ src, float* __restrict__ dst, int mode)
{
  int idx = blockIdx.x * 256 + threadIdx.x;
  if (idx >= DIM * NKP) return;
  int n = idx & (NKP - 1);
  int c = idx >> 10;
  int h = c & 3, hd = c >> 2;
  if (mode == 0)      dst[((size_t)(h * NKP + n)) * 64 + hd] = src[(size_t)c * NKP + n];
  else if (mode == 1) dst[((size_t)(h * 64 + hd)) * NKP + n] = src[(size_t)c * NKP + n];
  else                dst[(size_t)c * NKP + n] = src[(size_t)h * NKP * 64 + (size_t)n * 64 + hd];
}

// ---------------- row softmax over N columns ---------------------------------------------
__global__ void __launch_bounds__(256) softmax_rows_kernel(float* __restrict__ S, int ncols)
{
  float* p = S + (size_t)blockIdx.x * ncols;
  __shared__ float red[256];
  float mx = -3.4e38f;
  for (int j = threadIdx.x; j < ncols; j += 256) mx = fmaxf(mx, p[j]);
  red[threadIdx.x] = mx; __syncthreads();
  for (int s = 128; s > 0; s >>= 1) {
    if (threadIdx.x < s) red[threadIdx.x] = fmaxf(red[threadIdx.x], red[threadIdx.x + s]);
    __syncthreads();
  }
  mx = red[0]; __syncthreads();
  float sum = 0.f;
  for (int j = threadIdx.x; j < ncols; j += 256) { float e = expf(p[j] - mx); p[j] = e; sum += e; }
  red[threadIdx.x] = sum; __syncthreads();
  for (int s = 128; s > 0; s >>= 1) {
    if (threadIdx.x < s) red[threadIdx.x] += red[threadIdx.x + s];
    __syncthreads();
  }
  float inv = 1.f / red[0];
  for (int j = threadIdx.x; j < ncols; j += 256) p[j] *= inv;
}

// ---------------- generic transpose ------------------------------------------------------
__global__ void transpose_kernel(const float* __restrict__ src, float* __restrict__ dst,
                                 int rows, int cols)
{
  int idx = blockIdx.x * 256 + threadIdx.x;
  if (idx >= rows * cols) return;
  int r = idx / cols, c = idx % cols;
  dst[(size_t)c * rows + r] = src[(size_t)r * cols + c];
}

// ---------------- Sinkhorn ----------------------------------------------------------------
__global__ void build_couplings_kernel(const float* __restrict__ S, const float* __restrict__ alpha,
                                       float* __restrict__ Cc, float* __restrict__ uu,
                                       float* __restrict__ vv)
{
  int idx = blockIdx.x * 256 + threadIdx.x;
  if (idx < MP * MP) {
    int i = idx / MP, j = idx % MP;
    Cc[idx] = (i < NKP && j < NKP) ? S[(size_t)i * NKP + j] : alpha[0];
  }
  if (idx < MP) { uu[idx] = 0.f; vv[idx] = 0.f; }
}

__global__ void __launch_bounds__(256) sinkhorn_row_kernel(const float* __restrict__ Cc,
                                                           const float* __restrict__ vv,
                                                           float* __restrict__ uu)
{
  int i = blockIdx.x;
  const float* row = Cc + (size_t)i * MP;
  __shared__ float red[256];
  float mx = -3.4e38f;
  for (int j = threadIdx.x; j < MP; j += 256) mx = fmaxf(mx, row[j] + vv[j]);
  red[threadIdx.x] = mx; __syncthreads();
  for (int s = 128; s > 0; s >>= 1) {
    if (threadIdx.x < s) red[threadIdx.x] = fmaxf(red[threadIdx.x], red[threadIdx.x + s]);
    __syncthreads();
  }
  mx = red[0]; __syncthreads();
  float sum = 0.f;
  for (int j = threadIdx.x; j < MP; j += 256) sum += expf(row[j] + vv[j] - mx);
  red[threadIdx.x] = sum; __syncthreads();
  for (int s = 128; s > 0; s >>= 1) {
    if (threadIdx.x < s) red[threadIdx.x] += red[threadIdx.x + s];
    __syncthreads();
  }
  if (threadIdx.x == 0) {
    const float norm = -logf(2048.f);
    float logmu = (i < NKP) ? norm : (logf((float)NKP) + norm);
    uu[i] = logmu - (mx + logf(red[0]));
  }
}

__global__ void __launch_bounds__(256) sinkhorn_col_kernel(const float* __restrict__ Cc,
                                                           const float* __restrict__ uu,
                                                           float* __restrict__ vv)
{
  int j = blockIdx.x;
  __shared__ float red[256];
  float mx = -3.4e38f;
  for (int i = threadIdx.x; i < MP; i += 256) mx = fmaxf(mx, Cc[(size_t)i * MP + j] + uu[i]);
  red[threadIdx.x] = mx; __syncthreads();
  for (int s = 128; s > 0; s >>= 1) {
    if (threadIdx.x < s) red[threadIdx.x] = fmaxf(red[threadIdx.x], red[threadIdx.x + s]);
    __syncthreads();
  }
  mx = red[0]; __syncthreads();
  float sum = 0.f;
  for (int i = threadIdx.x; i < MP; i += 256) sum += expf(Cc[(size_t)i * MP + j] + uu[i] - mx);
  red[threadIdx.x] = sum; __syncthreads();
  for (int s = 128; s > 0; s >>= 1) {
    if (threadIdx.x < s) red[threadIdx.x] += red[threadIdx.x + s];
    __syncthreads();
  }
  if (threadIdx.x == 0) {
    const float norm = -logf(2048.f);
    float lognu = (j < NKP) ? norm : (logf((float)NKP) + norm);
    vv[j] = lognu - (mx + logf(red[0]));
  }
}

__global__ void finalize_Z_kernel(const float* __restrict__ Cc, const float* __restrict__ uu,
                                  const float* __restrict__ vv, float* __restrict__ Z)
{
  int idx = blockIdx.x * 256 + threadIdx.x;
  if (idx >= MP * MP) return;
  int i = idx / MP, j = idx % MP;
  Z[idx] = Cc[idx] + uu[i] + vv[j] + logf(2048.f);   // - norm, norm = -log(m+n)
}

// ---------------- matching ----------------------------------------------------------------
__global__ void __launch_bounds__(256) rowargmax_kernel(const float* __restrict__ Z,
                                                        float* __restrict__ mx, int* __restrict__ am)
{
  int i = blockIdx.x;
  const float* p = Z + (size_t)i * MP;
  float best = -3.4e38f; int bj = 0;
  for (int j = threadIdx.x; j < NKP; j += 256) {
    float v = p[j];
    if (v > best) { best = v; bj = j; }
  }
  __shared__ float sv[256]; __shared__ int si[256];
  sv[threadIdx.x] = best; si[threadIdx.x] = bj; __syncthreads();
  for (int s = 128; s > 0; s >>= 1) {
    if (threadIdx.x < s) {
      float ov = sv[threadIdx.x + s]; int oi = si[threadIdx.x + s];
      if (ov > sv[threadIdx.x] || (ov == sv[threadIdx.x] && oi < si[threadIdx.x])) {
        sv[threadIdx.x] = ov; si[threadIdx.x] = oi;
      }
    }
    __syncthreads();
  }
  if (threadIdx.x == 0) { mx[i] = sv[0]; am[i] = si[0]; }
}

__global__ void __launch_bounds__(256) colargmax_kernel(const float* __restrict__ Z,
                                                        float* __restrict__ mx, int* __restrict__ am)
{
  int j = blockIdx.x;
  float best = -3.4e38f; int bi = 0;
  for (int i = threadIdx.x; i < NKP; i += 256) {
    float v = Z[(size_t)i * MP + j];
    if (v > best) { best = v; bi = i; }
  }
  __shared__ float sv[256]; __shared__ int si[256];
  sv[threadIdx.x] = best; si[threadIdx.x] = bi; __syncthreads();
  for (int s = 128; s > 0; s >>= 1) {
    if (threadIdx.x < s) {
      float ov = sv[threadIdx.x + s]; int oi = si[threadIdx.x + s];
      if (ov > sv[threadIdx.x] || (ov == sv[threadIdx.x] && oi < si[threadIdx.x])) {
        sv[threadIdx.x] = ov; si[threadIdx.x] = oi;
      }
    }
    __syncthreads();
  }
  if (threadIdx.x == 0) { mx[j] = sv[0]; am[j] = si[0]; }
}

__global__ void __launch_bounds__(1024) match_kernel(
    const float* __restrict__ max0v, const int* __restrict__ idx0,
    const int* __restrict__ idx1,
    float* __restrict__ out_i0, float* __restrict__ out_i1,
    float* __restrict__ out_ms0, float* __restrict__ out_ms1,
    float* __restrict__ msc0ws, int* __restrict__ valid0ws)
{
  int n = threadIdx.x;
  int i0 = idx0[n];
  bool mutual0 = (idx1[i0] == n);
  float ms0 = mutual0 ? expf(max0v[n]) : 0.f;
  bool v0 = mutual0 && (ms0 > 0.2f);
  msc0ws[n] = ms0;
  valid0ws[n] = v0 ? 1 : 0;
  out_ms0[n] = ms0;
  out_i0[n] = v0 ? (float)i0 : -1.f;
  __syncthreads();
  int i1 = idx1[n];
  bool mutual1 = (idx0[i1] == n);
  float ms1 = mutual1 ? msc0ws[i1] : 0.f;
  bool v1 = mutual1 && (valid0ws[i1] != 0);
  out_ms1[n] = ms1;
  out_i1[n] = v1 ? (float)i1 : -1.f;
}

// =========================================================================================
extern "C" void kernel_launch(void* const* d_in, const int* in_sizes, int n_in,
                              void* d_out, int out_size, void* d_ws, size_t ws_size,
                              hipStream_t stream)
{
  (void)in_sizes; (void)n_in; (void)out_size; (void)ws_size;

  // ---- input map (recursive insertion order of setup_inputs dict) ----
  const float* desc0   = (const float*)d_in[0];
  const float* desc1   = (const float*)d_in[1];
  const float* kpts0   = (const float*)d_in[2];
  const float* kpts1   = (const float*)d_in[3];
  const float* scores0 = (const float*)d_in[4];
  const float* scores1 = (const float*)d_in[5];
  auto F = [&](int i) { return (const float*)d_in[i]; };
  // kenc layers: 0..3 -> {w,b,g,bb,m,v} at 6 + 6*i; layer 4 -> {w,b} at 30,31
  // gnn layer L base = 32 + 16*L:
  //   +0 qw +1 qb +2 kw +3 kb +4 vw +5 vb +6 mergew +7 mergeb
  //   +8 m0w +9 m0b +10 m0g +11 m0bb +12 m0m +13 m0v +14 m1w +15 m1b
  const float* final_w = F(320);
  const float* final_b = F(321);
  const float* bin_sc  = F(322);
  const int* wptr = (const int*)d_in[323];
  const int* hptr = (const int*)d_in[324];

  // ---- workspace layout (floats) ----
  float* W = (float*)d_ws;
  size_t off = 0;
  auto alloc = [&](size_t nf) { float* p = W + off; off += nf; return p; };
  float* bufCat = alloc(512 * NKP);      // kenc input rows / concat[x; msg]
  float* bufHid = alloc(512 * NKP);      // kenc stages / MLP hidden
  float* bufEnc = alloc(DIM * NKP);
  float* d0a = alloc(DIM * NKP); float* d0b = alloc(DIM * NKP);
  float* d1a = alloc(DIM * NKP); float* d1b = alloc(DIM * NKP);
  float* qb  = alloc(DIM * NKP); float* kb  = alloc(DIM * NKP); float* vb = alloc(DIM * NKP);
  float* Qt  = alloc(DIM * NKP); float* Kt  = alloc(DIM * NKP); float* Vt = alloc(DIM * NKP);
  float* Sb  = alloc((size_t)NKP * NKP); // per-head attention scores / final scores
  float* Ob  = alloc(DIM * NKP);         // per-head attention output [h][n][64]
  float* msg = alloc(DIM * NKP);
  float* md0 = alloc(DIM * NKP); float* md0t = alloc(DIM * NKP); float* md1 = alloc(DIM * NKP);
  float* Cc  = alloc((size_t)MP * MP);
  float* uu  = alloc(MP); float* vv = alloc(MP);
  float* max0v = alloc(NKP); float* max1v = alloc(NKP);
  int* idx0 = (int*)alloc(NKP); int* idx1 = (int*)alloc(NKP);
  float* msc0ws = alloc(NKP); int* valid0ws = (int*)alloc(NKP);

  auto gemm = [&](const float* A, int lda, const float* B, int ldb, float* C, int ldc,
                  int M, int N, int K, float alpha, const float* bias,
                  const float* g, const float* bb2, const float* mm, const float* vvv,
                  const float* resid, int ldr, int relu) {
    if ((N & 127) == 0) {
      dim3 grid(N / 128, (M + BM - 1) / BM);
      gemm_wmma_kernel<128><<<grid, 256, 0, stream>>>(A, lda, B, ldb, C, ldc, M, N, K, alpha,
                                                      bias, g, bb2, mm, vvv, resid, ldr, relu);
    } else {
      dim3 grid((N + 63) / 64, (M + BM - 1) / BM);
      gemm_wmma_kernel<64><<<grid, 256, 0, stream>>>(A, lda, B, ldb, C, ldc, M, N, K, alpha,
                                                     bias, g, bb2, mm, vvv, resid, ldr, relu);
    }
  };

  // ---- keypoint encoder for both images ----
  const int KC[6] = {3, 32, 64, 128, 256, 256};
  for (int img = 0; img < 2; ++img) {
    const float* kp = img ? kpts1 : kpts0;
    const float* sc = img ? scores1 : scores0;
    const float* dsc = img ? desc1 : desc0;
    float* dcur = img ? d1a : d0a;
    build_input_kernel<<<(NKP + 255) / 256, 256, 0, stream>>>(kp, sc, wptr, hptr, bufCat);
    const float* x = bufCat;
    float* pong[2] = {bufEnc, bufHid};
    for (int l = 0; l < 4; ++l) {
      int base = 6 + 6 * l;
      float* y = pong[l & 1];
      gemm(F(base + 0), KC[l], x, NKP, y, NKP, KC[l + 1], NKP, KC[l],
           1.f, F(base + 1), F(base + 2), F(base + 3), F(base + 4), F(base + 5),
           nullptr, 0, 1);
      x = y;
    }
    // last kenc conv + residual add of descriptor -> d0a/d1a
    gemm(F(30), 256, x, NKP, dcur, NKP, 256, NKP, 256,
         1.f, F(31), nullptr, nullptr, nullptr, nullptr, dsc, NKP, 0);
  }

  // ---- GNN: 18 layers, LAYER_NAMES = ['self','cross']*9 ----
  float* cur0 = d0a; float* cur1 = d1a; float* nxt0 = d0b; float* nxt1 = d1b;
  for (int L = 0; L < 18; ++L) {
    bool cross = (L & 1);
    int base = 32 + 16 * L;
    for (int img = 0; img < 2; ++img) {
      const float* x   = img ? cur1 : cur0;
      const float* src = cross ? (img ? cur0 : cur1) : x;
      float* dnew = img ? nxt1 : nxt0;

      gemm(F(base + 0), 256, x,   NKP, qb, NKP, 256, NKP, 256, 1.f, F(base + 1),
           nullptr, nullptr, nullptr, nullptr, nullptr, 0, 0);
      gemm(F(base + 2), 256, src, NKP, kb, NKP, 256, NKP, 256, 1.f, F(base + 3),
           nullptr, nullptr, nullptr, nullptr, nullptr, 0, 0);
      gemm(F(base + 4), 256, src, NKP, vb, NKP, 256, NKP, 256, 1.f, F(base + 5),
           nullptr, nullptr, nullptr, nullptr, nullptr, 0, 0);

      headshuffle_kernel<<<(DIM * NKP) / 256, 256, 0, stream>>>(qb, Qt, 0);
      headshuffle_kernel<<<(DIM * NKP) / 256, 256, 0, stream>>>(kb, Kt, 1);
      headshuffle_kernel<<<(DIM * NKP) / 256, 256, 0, stream>>>(vb, Vt, 0);

      for (int h = 0; h < 4; ++h) {
        const float* Qh = Qt + (size_t)h * NKP * 64;   // [n][64]
        const float* Kh = Kt + (size_t)h * 64 * NKP;   // [64][n]
        const float* Vh = Vt + (size_t)h * NKP * 64;   // [m][64]
        float* Oh = Ob + (size_t)h * NKP * 64;         // [n][64]
        gemm(Qh, 64, Kh, NKP, Sb, NKP, NKP, NKP, 64, 0.125f,
             nullptr, nullptr, nullptr, nullptr, nullptr, nullptr, 0, 0);
        softmax_rows_kernel<<<NKP, 256, 0, stream>>>(Sb, NKP);
        gemm(Sb, NKP, Vh, 64, Oh, 64, NKP, 64, NKP, 1.f,
             nullptr, nullptr, nullptr, nullptr, nullptr, nullptr, 0, 0);
      }
      headshuffle_kernel<<<(DIM * NKP) / 256, 256, 0, stream>>>(Ob, qb, 2);   // merged heads
      gemm(F(base + 6), 256, qb, NKP, msg, NKP, 256, NKP, 256, 1.f, F(base + 7),
           nullptr, nullptr, nullptr, nullptr, nullptr, 0, 0);

      // concat [x; message] -> 512 x N, then MLP
      hipMemcpyAsync(bufCat, x, (size_t)DIM * NKP * sizeof(float),
                     hipMemcpyDeviceToDevice, stream);
      hipMemcpyAsync(bufCat + (size_t)DIM * NKP, msg, (size_t)DIM * NKP * sizeof(float),
                     hipMemcpyDeviceToDevice, stream);
      gemm(F(base + 8), 512, bufCat, NKP, bufHid, NKP, 512, NKP, 512, 1.f, F(base + 9),
           F(base + 10), F(base + 11), F(base + 12), F(base + 13), nullptr, 0, 1);
      gemm(F(base + 14), 512, bufHid, NKP, dnew, NKP, 256, NKP, 512, 1.f, F(base + 15),
           nullptr, nullptr, nullptr, nullptr, x, NKP, 0);
    }
    float* t;
    t = cur0; cur0 = nxt0; nxt0 = t;
    t = cur1; cur1 = nxt1; nxt1 = t;
  }

  // ---- final projection + score matrix ----
  gemm(final_w, 256, cur0, NKP, md0, NKP, 256, NKP, 256, 1.f, final_b,
       nullptr, nullptr, nullptr, nullptr, nullptr, 0, 0);
  gemm(final_w, 256, cur1, NKP, md1, NKP, 256, NKP, 256, 1.f, final_b,
       nullptr, nullptr, nullptr, nullptr, nullptr, 0, 0);
  transpose_kernel<<<(DIM * NKP) / 256, 256, 0, stream>>>(md0, md0t, DIM, NKP);
  gemm(md0t, 256, md1, NKP, Sb, NKP, NKP, NKP, 256, 1.f / 16.f,
       nullptr, nullptr, nullptr, nullptr, nullptr, nullptr, 0, 0);

  // ---- Sinkhorn (100 iterations of row/col log-normalization) ----
  int nblk = (MP * MP + 255) / 256;
  build_couplings_kernel<<<nblk, 256, 0, stream>>>(Sb, bin_sc, Cc, uu, vv);
  for (int it = 0; it < 100; ++it) {
    sinkhorn_row_kernel<<<MP, 256, 0, stream>>>(Cc, vv, uu);
    sinkhorn_col_kernel<<<MP, 256, 0, stream>>>(Cc, uu, vv);
  }

  float* Zout = (float*)d_out;
  finalize_Z_kernel<<<nblk, 256, 0, stream>>>(Cc, uu, vv, Zout);

  // ---- matching ----
  rowargmax_kernel<<<NKP, 256, 0, stream>>>(Zout, max0v, idx0);
  colargmax_kernel<<<NKP, 256, 0, stream>>>(Zout, max1v, idx1);
  float* out_i0  = Zout + (size_t)MP * MP;
  float* out_i1  = out_i0 + NKP;
  float* out_ms0 = out_i1 + NKP;
  float* out_ms1 = out_ms0 + NKP;
  match_kernel<<<1, 1024, 0, stream>>>(max0v, idx0, idx1,
                                       out_i0, out_i1, out_ms0, out_ms1,
                                       msc0ws, valid0ws);
}
